// MultiHeadedAttention_57784490000738
// MI455X (gfx1250) — matrix operand order repaired and checked by reading, not compiled, not run on hardware
//
#include <hip/hip_runtime.h>

#define NUM_HEADS 16
#define EMBED     1024
#define HD        64
#define BATCH     2
#define SEQ       2048
#define MTOT      (BATCH * SEQ)   // 4096 token rows

typedef __attribute__((ext_vector_type(16))) __bf16 b16x16;
typedef __attribute__((ext_vector_type(8)))  __bf16 b16x8;
typedef __attribute__((ext_vector_type(8)))  float  f32x8;

// ---- fragment helpers (CDNA5 WMMA 16x16x32 bf16 fragments, contiguous-K order of this code) ----
static __device__ __forceinline__ b16x16 join8(b16x8 lo, b16x8 hi) {
  return __builtin_shufflevector(lo, hi, 0, 1, 2, 3, 4, 5, 6, 7,
                                         8, 9, 10, 11, 12, 13, 14, 15);
}
// A 16x32: lane holds row (lane&15); K = 0..15 (lanes<16) or
// 16..31 (lanes>=16).  p must already include +half*16.
static __device__ __forceinline__ b16x16 loadA(const __bf16* p) {
  return join8(*(const b16x8*)p, *(const b16x8*)(p + 8));
}
// B 32x16: lane holds column (lane&15); K = 0..15 (lanes<16) or 16..31.
// p must already include +half*16; 16 contiguous bf16.
static __device__ __forceinline__ b16x16 loadB(const __bf16* p) {
  return join8(*(const b16x8*)p, *(const b16x8*)(p + 8));
}
static __device__ __forceinline__ f32x8 wmma_bf16(b16x16 a, b16x16 b, f32x8 c) {
  return __builtin_amdgcn_wmma_f32_16x16x32_bf16(false, a, false, b,
                                                 (short)0, c, false, false);
}

// gfx1250 async DMA: global -> LDS, 16 bytes per lane, tracked by ASYNCcnt.
static __device__ __forceinline__ void async_b128(unsigned lds_off, const void* gptr) {
  asm volatile("global_load_async_to_lds_b128 %0, %1, off"
               :: "v"(lds_off), "v"((unsigned long long)(uintptr_t)gptr)
               : "memory");
}

// ---- conversion / layout kernels ----
__global__ void f32_to_bf16(const float* __restrict__ in, __bf16* __restrict__ out, int n) {
  int i = blockIdx.x * 256 + threadIdx.x;
  if (i < n) out[i] = (__bf16)in[i];
}

// W [K=1024][N=1024] fp32 -> Wt [N][K] bf16 (so B-columns are contiguous)
__global__ void transpose_w(const float* __restrict__ W, __bf16* __restrict__ Wt) {
  int i = blockIdx.x * 256 + threadIdx.x;       // coalesced read over 1Mi
  int n = i & 1023, k = i >> 10;
  Wt[(size_t)n * EMBED + k] = (__bf16)W[i];
}

// [b,s,h,d] (row-major 4096x1024) -> [b,h,s,d]
__global__ void reshape_qk(const __bf16* __restrict__ in, __bf16* __restrict__ out) {
  int i = blockIdx.x * 256 + threadIdx.x;       // 0 .. 2^22-1
  int d = i & 63, h = (i >> 6) & 15, s = (i >> 10) & 2047, b = i >> 21;
  out[((size_t)(b * NUM_HEADS + h) * SEQ + s) * HD + d] = in[i];
}

// [b,s,h,d] -> [b,h,d,s]  (V transposed so PV B-fragments are contiguous)
__global__ void reshape_vt(const __bf16* __restrict__ in, __bf16* __restrict__ out) {
  int i = blockIdx.x * 256 + threadIdx.x;
  int d = i & 63, h = (i >> 6) & 15, s = (i >> 10) & 2047, b = i >> 21;
  out[((size_t)(b * NUM_HEADS + h) * HD + d) * SEQ + s] = in[i];
}

// ---- GEMM: C[4096][1024] = A(bf16) * Bt(bf16)^T + bias ----
// Block tile 128x128, double-buffered LDS staging via async DMA.
// 8 waves: 4 along M (32 rows each) x 2 along N (64 cols each); 8 WMMA / k-step.
__global__ __launch_bounds__(256)
void gemm_bf16(const __bf16* __restrict__ A, const __bf16* __restrict__ Bt,
               const float* __restrict__ bias, void* __restrict__ outp, int out_f32) {
  __shared__ __bf16 ldsA[2 * 128 * 32];   // [buf][row][k]  64B rows
  __shared__ __bf16 ldsB[2 * 128 * 32];   // [buf][col][k]
  const int lane = threadIdx.x & 31;
  const int wave = threadIdx.x >> 5;
  const int half = lane >> 4, l16 = lane & 15;
  const int mbase = blockIdx.x * 128;
  const int nbase = blockIdx.y * 128;
  const int mrow = (wave & 3) * 32;       // wave's M offset in block tile
  const int ncol = (wave >> 2) * 64;      // wave's N offset in block tile

  const unsigned ldsA_base = (unsigned)(uintptr_t)ldsA;
  const unsigned ldsB_base = (unsigned)(uintptr_t)ldsB;
  // staging assignment: wave stages rows wave*16 .. wave*16+15 of both tiles,
  // 4 lanes per 64B row, 16B per lane
  const int srow = wave * 16 + (lane >> 2);
  const int sbyte = (lane & 3) * 16;      // byte offset within row
  const int selem = (lane & 3) * 8;       // element offset within row

  f32x8 acc[2][4] = {};

  // prologue: stage k0 = 0 into buffer 0
  {
    const __bf16* ga = A  + (size_t)(mbase + srow) * EMBED + selem;
    const __bf16* gb = Bt + (size_t)(nbase + srow) * EMBED + selem;
    unsigned la = ldsA_base + srow * 64 + sbyte;
    unsigned lb = ldsB_base + srow * 64 + sbyte;
    async_b128(la, ga);            async_b128(la + 8 * 64, ga + 8 * EMBED);
    async_b128(lb, gb);            async_b128(lb + 8 * 64, gb + 8 * EMBED);
  }

  int cur = 0;
  for (int k0 = 0; k0 < EMBED; k0 += 32, cur ^= 1) {
    const int knext = k0 + 32;
    if (knext < EMBED) {
      const int nb = cur ^ 1;
      const __bf16* ga = A  + (size_t)(mbase + srow) * EMBED + knext + selem;
      const __bf16* gb = Bt + (size_t)(nbase + srow) * EMBED + knext + selem;
      unsigned la = ldsA_base + nb * 8192 + srow * 64 + sbyte;
      unsigned lb = ldsB_base + nb * 8192 + srow * 64 + sbyte;
      async_b128(la, ga);          async_b128(la + 8 * 64, ga + 8 * EMBED);
      async_b128(lb, gb);          async_b128(lb + 8 * 64, gb + 8 * EMBED);
      asm volatile("s_wait_asynccnt 0x4" ::: "memory");  // current tile's 4 done
    } else {
      asm volatile("s_wait_asynccnt 0x0" ::: "memory");
    }
    __syncthreads();                       // all waves' DMA for buf[cur] visible

    const __bf16* la = &ldsA[cur * 4096];
    const __bf16* lb = &ldsB[cur * 4096];
    b16x16 a0 = loadA(la + (mrow +      l16) * 32 + half * 16);
    b16x16 a1 = loadA(la + (mrow + 16 + l16) * 32 + half * 16);
#pragma unroll
    for (int c = 0; c < 4; ++c) {
      b16x16 bf = loadB(lb + (ncol + c * 16 + l16) * 32 + half * 16);
      acc[0][c] = wmma_bf16(a0, bf, acc[0][c]);
      acc[1][c] = wmma_bf16(a1, bf, acc[1][c]);
    }
    __syncthreads();                       // done reading buf[cur] before overwrite
  }

#pragma unroll
  for (int i = 0; i < 2; ++i)
#pragma unroll
    for (int c = 0; c < 4; ++c) {
      int col = nbase + ncol + c * 16 + l16;
      float bs = bias[col];
#pragma unroll
      for (int r = 0; r < 8; ++r) {
        int row = mbase + mrow + i * 16 + r + 8 * half;  // C layout: M = r + 8*half
        float v = acc[i][c][r] + bs;
        if (out_f32) ((float*)outp)[(size_t)row * EMBED + col] = v;
        else         ((__bf16*)outp)[(size_t)row * EMBED + col] = (__bf16)v;
      }
    }
}

// ---- flash attention: one wave per 16-query tile, 32 keys per step ----
__global__ __launch_bounds__(128)
void attn_fused(const __bf16* __restrict__ Q, const __bf16* __restrict__ K,
                const __bf16* __restrict__ Vt, __bf16* __restrict__ O) {
  __shared__ __bf16 ldsP[4][16 * 32];            // per-wave P tile (16q x 32k)
  const int lane = threadIdx.x & 31;
  const int wave = threadIdx.x >> 5;
  const int half = lane >> 4, l16 = lane & 15;
  const int tile = blockIdx.x * 4 + wave;        // 0..4095 q-tiles
  const int qt = tile & (SEQ / 16 - 1);
  const int bh = tile >> 7;                      // /(SEQ/16)
  const __bf16* qb = Q  + (size_t)bh * SEQ * HD;
  const __bf16* kb = K  + (size_t)bh * SEQ * HD;
  const __bf16* vb = Vt + (size_t)bh * HD * SEQ;

  // Q A-fragments for d = 0..31 and 32..63
  b16x16 aq0 = loadA(qb + (size_t)(qt * 16 + l16) * HD + 0  + half * 16);
  b16x16 aq1 = loadA(qb + (size_t)(qt * 16 + l16) * HD + 32 + half * 16);

  float m[8], l[8];
  f32x8 acc[4] = {};
#pragma unroll
  for (int r = 0; r < 8; ++r) { m[r] = -3.0e38f; l[r] = 0.f; }

  for (int j0 = 0; j0 < SEQ; j0 += 32) {
    // two 16-key score subtiles: S = Q(16x64) @ K^T, contraction over d
    f32x8 sc[2];
#pragma unroll
    for (int sub = 0; sub < 2; ++sub) {
      const __bf16* kp = kb + (size_t)(j0 + sub * 16 + l16) * HD + half * 16;
      f32x8 s0 = {};
      s0 = wmma_bf16(aq0, loadB(kp), s0);
      s0 = wmma_bf16(aq1, loadB(kp + 32), s0);
      sc[sub] = s0;
    }
    const float scale = 0.125f;                  // 1/sqrt(64)
    float p0[8], p1[8], corr[8];
#pragma unroll
    for (int r = 0; r < 8; ++r) {
      float x0 = sc[0][r] * scale, x1 = sc[1][r] * scale;
      float cm = fmaxf(x0, x1);                  // row max across 16 lanes (same half)
#pragma unroll
      for (int s = 1; s < 16; s <<= 1) cm = fmaxf(cm, __shfl_xor(cm, s, 32));
      float nm = fmaxf(m[r], cm);
      corr[r] = __expf(m[r] - nm);
      p0[r] = __expf(x0 - nm);
      p1[r] = __expf(x1 - nm);
      float rs = p0[r] + p1[r];
#pragma unroll
      for (int s = 1; s < 16; s <<= 1) rs += __shfl_xor(rs, s, 32);
      l[r] = l[r] * corr[r] + rs;
      m[r] = nm;
    }
    // C-layout -> A-layout transpose of P via LDS
#pragma unroll
    for (int r = 0; r < 8; ++r) {
      int row = r + 8 * half;
      ldsP[wave][row * 32 + l16]      = (__bf16)p0[r];
      ldsP[wave][row * 32 + 16 + l16] = (__bf16)p1[r];
    }
#pragma unroll
    for (int c = 0; c < 4; ++c)
#pragma unroll
      for (int r = 0; r < 8; ++r) acc[c][r] *= corr[r];
    __syncthreads();
    b16x16 aP = loadA(&ldsP[wave][l16 * 32 + half * 16]);
#pragma unroll
    for (int c = 0; c < 4; ++c) {                // O(16x64) += P(16x32) @ V(32x16)
      const __bf16* vp = vb + (size_t)(c * 16 + l16) * SEQ + j0 + half * 16;
      acc[c] = wmma_bf16(aP, loadB(vp), acc[c]);
    }
    __syncthreads();
  }
  const int b = bh >> 4, h = bh & 15;
#pragma unroll
  for (int c = 0; c < 4; ++c) {
    int col = h * HD + c * 16 + l16;
#pragma unroll
    for (int r = 0; r < 8; ++r) {
      int srow = qt * 16 + r + 8 * half;
      O[((size_t)(b * SEQ + srow)) * EMBED + col] = (__bf16)(acc[c][r] / l[r]);
    }
  }
}

extern "C" void kernel_launch(void* const* d_in, const int* in_sizes, int n_in,
                              void* d_out, int out_size, void* d_ws, size_t ws_size,
                              hipStream_t stream) {
  const float* x  = (const float*)d_in[0];
  const float* Wq = (const float*)d_in[1];
  const float* bq = (const float*)d_in[2];
  const float* Wk = (const float*)d_in[3];
  const float* bk = (const float*)d_in[4];
  const float* Wv = (const float*)d_in[5];
  const float* bv = (const float*)d_in[6];
  const float* Wo = (const float*)d_in[7];
  const float* bo = (const float*)d_in[8];

  const size_t NX = (size_t)MTOT * EMBED;   // 4 Mi elements
  const size_t NW = (size_t)EMBED * EMBED;  // 1 Mi elements
  char* w = (char*)d_ws;
  __bf16* xb   = (__bf16*)w; w += NX * 2;
  __bf16* wqt  = (__bf16*)w; w += NW * 2;
  __bf16* wkt  = (__bf16*)w; w += NW * 2;
  __bf16* wvt  = (__bf16*)w; w += NW * 2;
  __bf16* wot  = (__bf16*)w; w += NW * 2;
  __bf16* tmp  = (__bf16*)w; w += NX * 2;
  __bf16* qbuf = (__bf16*)w; w += NX * 2;
  __bf16* kbuf = (__bf16*)w; w += NX * 2;
  __bf16* vtbf = (__bf16*)w; w += NX * 2;
  __bf16* aout = (__bf16*)w; w += NX * 2;   // ~56 MB total

  f32_to_bf16<<<(int)(NX / 256), 256, 0, stream>>>(x, xb, (int)NX);
  transpose_w<<<(int)(NW / 256), 256, 0, stream>>>(Wq, wqt);
  transpose_w<<<(int)(NW / 256), 256, 0, stream>>>(Wk, wkt);
  transpose_w<<<(int)(NW / 256), 256, 0, stream>>>(Wv, wvt);
  transpose_w<<<(int)(NW / 256), 256, 0, stream>>>(Wo, wot);

  dim3 ggrid(MTOT / 128, EMBED / 128);
  gemm_bf16<<<ggrid, 256, 0, stream>>>(xb, wqt, bq, tmp, 0);
  reshape_qk<<<(int)(NX / 256), 256, 0, stream>>>(tmp, qbuf);
  gemm_bf16<<<ggrid, 256, 0, stream>>>(xb, wkt, bk, tmp, 0);
  reshape_qk<<<(int)(NX / 256), 256, 0, stream>>>(tmp, kbuf);
  gemm_bf16<<<ggrid, 256, 0, stream>>>(xb, wvt, bv, tmp, 0);
  reshape_vt<<<(int)(NX / 256), 256, 0, stream>>>(tmp, vtbf);

  attn_fused<<<(BATCH * NUM_HEADS * SEQ / 16) / 4, 128, 0, stream>>>(qbuf, kbuf, vtbf, aout);

  gemm_bf16<<<ggrid, 256, 0, stream>>>(aout, wot, bo, d_out, 1);

  (void)in_sizes; (void)n_in; (void)out_size; (void)ws_size;
}
